// S4D_35751307771909
// MI455X (gfx1250) — matrix operand order repaired and checked by reading, not compiled, not run on hardware
//
#include <hip/hip_runtime.h>
#include <hip/hip_bf16.h>
#include <math.h>

#define H_ 512
#define N_ 32
#define B_ 4
#define L_ 2048
#define M_TOT (B_ * L_)   // 8192 rows for the GEMM

typedef __attribute__((ext_vector_type(8)))  float  v8f;
typedef __attribute__((ext_vector_type(8)))  __bf16 v8bf;
typedef __attribute__((ext_vector_type(16))) __bf16 v16bf;

// ---- workspace layout (bytes) ----
// [0,            8388608)  Y bf16, row-major (M_TOT x H_)
// [8388608,      8912896)  W1 packed bf16 B-fragments
// [8912896,      9437184)  W2 packed bf16 B-fragments
// [9437184,      9830400)  6 x (H_*N_) f32: AdR AdI BdR BdI CR CI
#define OFF_Y   0u
#define OFF_W1P 8388608u
#define OFF_W2P 8912896u
#define OFF_PRM 9437184u
#define HN (H_ * N_)

__device__ __forceinline__ unsigned short f2bf(float f) {
  union { float f; unsigned u; } v; v.f = f;
  unsigned u = v.u;
  u += 0x7FFFu + ((u >> 16) & 1u);            // round-to-nearest-even
  return (unsigned short)(u >> 16);
}

// ---------------------------------------------------------------------------
// 1) Discretization: Ad = (1+dtA/2)/(1-dtA/2), Bd = dt*Bc/(1-dtA/2), C
// ---------------------------------------------------------------------------
__global__ void s4d_precompute(const float* __restrict__ log_dt,
                               const float* __restrict__ log_A_real,
                               const float* __restrict__ A_imag,
                               const float* __restrict__ B_ri,
                               const float* __restrict__ C_ri,
                               float* __restrict__ AdR, float* __restrict__ AdI,
                               float* __restrict__ BdR, float* __restrict__ BdI,
                               float* __restrict__ CR,  float* __restrict__ CI) {
  int i = blockIdx.x * blockDim.x + threadIdx.x;
  if (i >= HN) return;
  int h = i / N_, n = i % N_;
  float dt  = __expf(log_dt[h]);
  float Ar  = -__expf(log_A_real[i]);
  float Ai  = A_imag[i];
  float tr  = 0.5f * dt * Ar;                 // dtA real
  float ti  = 0.5f * dt * Ai;                 // dtA imag
  float dr  = 1.0f - tr, di = -ti;            // denom = 1 - dtA
  float den = dr * dr + di * di;
  float inv = 1.0f / den;
  float nr  = 1.0f + tr, ni = ti;             // numer = 1 + dtA
  AdR[i] = (nr * dr + ni * di) * inv;
  AdI[i] = (ni * dr - nr * di) * inv;
  float br = dt * B_ri[2 * n], bi = dt * B_ri[2 * n + 1];
  BdR[i] = (br * dr + bi * di) * inv;
  BdI[i] = (bi * dr - br * di) * inv;
  CR[i]  = C_ri[2 * i];
  CI[i]  = C_ri[2 * i + 1];
}

// ---------------------------------------------------------------------------
// 2) Pack W (f32 row-major HxH) into bf16 WMMA B-fragment order.
//    B order used here, k contiguous (32x16): lanes 0-15 hold K=0..15, lanes 16-31 hold
//    K=16..31; N = lane%16. Lane's 16 elems stored contiguously (32B).
//    idx bits: e[3:0] lane[8:4] kt[12:9] nTile[17:13]
// ---------------------------------------------------------------------------
__global__ void s4d_pack_w(const float* __restrict__ W,
                           unsigned short* __restrict__ Wp) {
  int idx = blockIdx.x * blockDim.x + threadIdx.x;
  if (idx >= (H_ / 16) * (H_ / 32) * 32 * 16) return;  // 262144
  int e        = idx & 15;
  int lane     = (idx >> 4) & 31;
  int kt       = (idx >> 9) & 15;
  int nTile    = idx >> 13;
  int laneHalf = lane >> 4, nn = lane & 15;
  int k = kt * 32 + laneHalf * 16 + e;
  Wp[idx] = f2bf(W[(size_t)k * H_ + nTile * 16 + nn]);
}

// ---------------------------------------------------------------------------
// 3) Scan: one wave32 per (b,h) sequence, lane n owns complex state n.
//    32 timesteps per global load/store block via lane shuffles; the C.x
//    reduction (5x shfl_xor) is off the serial xr/xi dependence chain.
// ---------------------------------------------------------------------------
__global__ __launch_bounds__(256)
void s4d_scan(const float* __restrict__ x,
              const float* __restrict__ AdR, const float* __restrict__ AdI,
              const float* __restrict__ BdR, const float* __restrict__ BdI,
              const float* __restrict__ CR,  const float* __restrict__ CI,
              const float* __restrict__ Dv,
              unsigned short* __restrict__ Ybf) {
  int wave = (blockIdx.x * blockDim.x + threadIdx.x) >> 5;
  int lane = threadIdx.x & 31;
  if (wave >= B_ * H_) return;
  int b = wave / H_, h = wave % H_;
  int pi = h * N_ + lane;
  float adr = AdR[pi], adi = AdI[pi];
  float bdr = BdR[pi], bdi = BdI[pi];
  float cr  = CR[pi],  ci  = CI[pi];
  float Dh  = Dv[h];
  float xr = 0.0f, xi = 0.0f;
  const float* xin = x + ((size_t)b * L_) * H_ + h;
  unsigned short* yout = Ybf + ((size_t)b * L_) * H_ + h;

  for (int l0 = 0; l0 < L_; l0 += 32) {
    float uvec = xin[(size_t)(l0 + lane) * H_];   // 32 timesteps, 1 per lane
    float myY = 0.0f;
    for (int t = 0; t < 32; ++t) {
      float u = __shfl(uvec, t, 32);
      float nxr = adr * xr - adi * xi + bdr * u;  // serial chain: 2 cplx FMAs
      float nxi = adi * xr + adr * xi + bdi * u;
      xr = nxr; xi = nxi;
      float c = cr * xr - ci * xi;                // Re(C * x), off-chain
      c += __shfl_xor(c, 16, 32);
      c += __shfl_xor(c,  8, 32);
      c += __shfl_xor(c,  4, 32);
      c += __shfl_xor(c,  2, 32);
      c += __shfl_xor(c,  1, 32);
      float y = c + Dh * u;
      if (lane == t) myY = y;                     // lane t keeps timestep t
    }
    yout[(size_t)(l0 + lane) * H_] = f2bf(myY);
  }
}

// ---------------------------------------------------------------------------
// 4) Fused GLU GEMM: out = (Y@W1) * sigmoid(Y@W2), bf16 WMMA, f32 accum.
//    All 8 waves in a block share one nGroup, so the B-fragments (8KB per
//    k-step: 4 nTiles x {W1,W2}) are staged ONCE per block into LDS with
//    double-buffered global_load_async_to_lds_b128 (ASYNCcnt), then read by
//    every wave via ds_load. A-fragments stream from global per wave.
// ---------------------------------------------------------------------------
__global__ __launch_bounds__(256)
void s4d_gemm_glu(const __bf16* __restrict__ Y,
                  const __bf16* __restrict__ W1p,
                  const __bf16* __restrict__ W2p,
                  float* __restrict__ out) {
  __shared__ __align__(32) __bf16 bstage[2][4096];   // 2 x 8KB double buffer

  int tid       = threadIdx.x;
  int lane      = tid & 31;
  int waveInBlk = tid >> 5;
  int blk       = blockIdx.x;                   // 512 blocks
  int nGroup    = blk >> 6;                     // 8 n-groups of 64 cols
  int mTile     = (blk & 63) * 8 + waveInBlk;   // 512 m-tiles of 16 rows
  int laneHalf  = lane >> 4, laneN = lane & 15;

  // Staging role: chunk = waveInBlk -> (jstage = chunk>>1, mat = chunk&1).
  // Each thread copies 32B of one 1KB fragment per k-step (2 async b128).
  const __bf16* wsrc = (waveInBlk & 1) ? W2p : W1p;
  int jstage = waveInBlk >> 1;
  const __bf16* srcBase =
      wsrc + ((((size_t)(nGroup * 4 + jstage)) * 16) * 32 + lane) * 16;
  // raw LDS offset: low 32 bits of the generic (flat) shared address
  unsigned ldsBase = (unsigned)(size_t)(void*)&bstage[0][0];
  unsigned dstBase = ldsBase + (unsigned)waveInBlk * 1024u
                             + (unsigned)lane * 32u;

  // prologue: stage kt = 0 into buffer 0
  {
    const __bf16* s0 = srcBase;
    asm volatile("global_load_async_to_lds_b128 %0, %1, off"
                 :: "v"(dstBase), "v"(s0) : "memory");
    asm volatile("global_load_async_to_lds_b128 %0, %1, off offset:16"
                 :: "v"(dstBase), "v"(s0) : "memory");
  }

  v8f acc1[4], acc2[4];
#pragma unroll
  for (int j = 0; j < 4; ++j) {
    acc1[j] = (v8f){0.f, 0.f, 0.f, 0.f, 0.f, 0.f, 0.f, 0.f};
    acc2[j] = acc1[j];
  }

  int row = mTile * 16 + laneN;
  const __bf16* arowp = Y + (size_t)row * H_ + laneHalf * 16;

  for (int kt = 0; kt < 16; ++kt) {
    int buf = kt & 1;
    asm volatile("s_wait_asynccnt 0x0" ::: "memory");  // my staged loads done
    __syncthreads();                                   // visible block-wide

    if (kt + 1 < 16) {                                 // stage next k-step
      const __bf16* s = srcBase + (size_t)(kt + 1) * 512;
      unsigned d = dstBase + (unsigned)(buf ^ 1) * 8192u;
      asm volatile("global_load_async_to_lds_b128 %0, %1, off"
                   :: "v"(d), "v"(s) : "memory");
      asm volatile("global_load_async_to_lds_b128 %0, %1, off offset:16"
                   :: "v"(d), "v"(s) : "memory");
    }

    // A-frag (16x32 bf16, matches packed B): lane<16 -> K 0..15,
    // lane>=16 -> K 16..31; two contiguous 16B global loads.
    const __bf16* pa = arowp + kt * 32;
    if (kt + 1 < 16)
      __builtin_prefetch(pa + 32, 0, 3);               // global_prefetch_b8
    v8bf alo = *(const v8bf*)pa;
    v8bf ahi = *(const v8bf*)(pa + 8);
    v16bf a = __builtin_shufflevector(alo, ahi, 0, 1, 2, 3, 4, 5, 6, 7,
                                      8, 9, 10, 11, 12, 13, 14, 15);

    const __bf16* lb = &bstage[buf][0];
#pragma unroll
    for (int j = 0; j < 4; ++j) {
      v16bf b1 = *(const v16bf*)(lb + (j * 2 + 0) * 512 + lane * 16);
      v16bf b2 = *(const v16bf*)(lb + (j * 2 + 1) * 512 + lane * 16);
      acc1[j] = __builtin_amdgcn_wmma_f32_16x16x32_bf16(
          false, a, false, b1, (short)0, acc1[j], false, false);
      acc2[j] = __builtin_amdgcn_wmma_f32_16x16x32_bf16(
          false, a, false, b2, (short)0, acc2[j], false, false);
    }
    __syncthreads();           // all waves done with buf before re-staging it
  }

  // C/D layout: VGPR r, lanes 0-15 -> M=r, lanes 16-31 -> M=r+8; N=lane%16
  int orow = mTile * 16 + laneHalf * 8;
#pragma unroll
  for (int j = 0; j < 4; ++j) {
    int col = nGroup * 64 + j * 16 + laneN;
#pragma unroll
    for (int r = 0; r < 8; ++r) {
      float g = acc1[j][r], s = acc2[j][r];
      float val = g / (1.0f + __expf(-s));
      out[(size_t)(orow + r) * H_ + col] = val;
    }
  }
}

// ---------------------------------------------------------------------------
extern "C" void kernel_launch(void* const* d_in, const int* in_sizes, int n_in,
                              void* d_out, int out_size, void* d_ws, size_t ws_size,
                              hipStream_t stream) {
  const float* x      = (const float*)d_in[0];
  const float* log_dt = (const float*)d_in[1];
  const float* lAr    = (const float*)d_in[2];
  const float* Aim    = (const float*)d_in[3];
  const float* Bri    = (const float*)d_in[4];
  const float* Cri    = (const float*)d_in[5];
  const float* Dv     = (const float*)d_in[6];
  const float* W1     = (const float*)d_in[7];
  const float* W2     = (const float*)d_in[8];
  float* out = (float*)d_out;

  char* ws = (char*)d_ws;
  unsigned short* Ybf = (unsigned short*)(ws + OFF_Y);
  unsigned short* W1p = (unsigned short*)(ws + OFF_W1P);
  unsigned short* W2p = (unsigned short*)(ws + OFF_W2P);
  float* P = (float*)(ws + OFF_PRM);
  float* AdR = P;            float* AdI = P + HN;
  float* BdR = P + 2 * HN;   float* BdI = P + 3 * HN;
  float* CR  = P + 4 * HN;   float* CI  = P + 5 * HN;

  s4d_precompute<<<(HN + 255) / 256, 256, 0, stream>>>(
      log_dt, lAr, Aim, Bri, Cri, AdR, AdI, BdR, BdI, CR, CI);
  s4d_pack_w<<<1024, 256, 0, stream>>>(W1, W1p);
  s4d_pack_w<<<1024, 256, 0, stream>>>(W2, W2p);
  s4d_scan<<<(B_ * H_ * 32) / 256, 256, 0, stream>>>(
      x, AdR, AdI, BdR, BdI, CR, CI, Dv, Ybf);
  s4d_gemm_glu<<<512, 256, 0, stream>>>(
      (const __bf16*)Ybf, (const __bf16*)W1p, (const __bf16*)W2p, out);
}
